// PGJANET_48988396978778
// MI455X (gfx1250) — compile-verified, matrix operand order checked by reading, not run on hardware
//
#include <hip/hip_runtime.h>
#include <hip/hip_bf16.h>

// PGJANET on MI455X (gfx1250): persistent per-batch-tile kernel, bf16 WMMA.
// B=128,T=1024,H=256,OUT=2. 8 WGs x 512 threads (16 wave32 waves).
// Each wave owns one 16-column N-tile of H. Weights pre-packed into WMMA
// B-fragment layout (ISA 7.12.2) in bf16. Register file holds what fits
// (compiler-promoted, loop-invariant); a 112KB slice of Wz is TDM-staged
// into LDS; the remainder streams from L2.

#define BB   128
#define TT   1024
#define HH   256
#define P32  257   // padded f32 h row stride (bank-conflict free)
#define P16  264   // padded bf16 row stride (16B multiple, conflict free)

typedef unsigned int   uint32;
typedef unsigned short ushort16;
typedef __bf16 v16bf __attribute__((ext_vector_type(16)));
typedef float  v8f   __attribute__((ext_vector_type(8)));
typedef unsigned int u32x4 __attribute__((ext_vector_type(4)));
typedef int          i32x8 __attribute__((ext_vector_type(8)));
typedef int          i32x4 __attribute__((ext_vector_type(4)));

// aux (f32) layout offsets
#define AX_WA0  0
#define AX_WP10 256
#define AX_WP20 512
#define AX_BA   768
#define AX_BP1  1024
#define AX_BP2  1280
#define AX_BZ   1536
#define AX_BH   1792
#define AX_WO   2048
#define AX_BO   2560
#define AX_N    2562

#define S1_HALVES (3*16*8*32*16)    // 196608
#define S2_HALVES (2*16*16*32*16)   // 262144

#define WZ_LDS_CHUNKS 7             // K-chunks of Wz cached in LDS (per ntile)

__device__ __forceinline__ ushort16 f2bf(float f) {
  union { float f; uint32 u; } v; v.f = f;
  uint32 r = v.u + 0x7FFFu + ((v.u >> 16) & 1u);   // round-to-nearest-even
  return (ushort16)(r >> 16);
}

__device__ __forceinline__ float fast_tanh(float x) {
  float r;
  // CDNA5 transcendental; s_delay_alu covers the TRANS32 RAW hazard.
  asm volatile("v_tanh_f32 %0, %1\n\ts_delay_alu instid0(TRANS32_DEP_1)"
               : "=v"(r) : "v"(x));
  return r;
}

__device__ __forceinline__ v8f wmma_bf16(v16bf a, v16bf b, v8f c) {
  return __builtin_amdgcn_wmma_f32_16x16x32_bf16(
      /*neg_a=*/false, a, /*neg_b=*/false, b,
      /*c_mod=*/(short)0, c, /*reuse_a=*/false, /*reuse_b=*/false);
}

// A-fragment per ISA 16-bit 16x32 layout: lane m=l&15; lanes<16 hold K
// {0..7,16..23}+32c, lanes>=16 hold K {8..15,24..31}+32c.
__device__ __forceinline__ v16bf load_afrag(const ushort16* row, int k0) {
  v16bf a;
  ((uint4*)&a)[0] = *(const uint4*)(row + k0);
  ((uint4*)&a)[1] = *(const uint4*)(row + k0 + 16);
  return a;
}

__device__ __forceinline__ v16bf load_bfrag(const ushort16* p) {
  v16bf b;
  ((uint4*)&b)[0] = ((const uint4*)p)[0];
  ((uint4*)&b)[1] = ((const uint4*)p)[1];
  return b;
}

// ---------------- weight packing: fragment-contiguous bf16 ----------------
// pk1: [mat(3)][ntile(16)][kchunk(8)][lane(32)][i(16)]  (Wa/Wp1/Wp2 rows 1..256)
// pk2: [mat(2)][ntile(16)][kchunk(16)][lane(32)][i(16)] (Wz/Wh, K=512)
__global__ void pgj_pack_weights(const float* __restrict__ Wa,
                                 const float* __restrict__ Wp1,
                                 const float* __restrict__ Wp2,
                                 const float* __restrict__ Wz,
                                 const float* __restrict__ Wh,
                                 ushort16* __restrict__ pk1,
                                 ushort16* __restrict__ pk2) {
  int idx = blockIdx.x * blockDim.x + threadIdx.x;
  if (idx < S1_HALVES) {
    int i  =  idx        & 15;
    int l  = (idx >> 4)  & 31;
    int c  = (idx >> 9)  & 7;
    int nt = (idx >> 12) & 15;
    int mat = idx >> 16;
    int n = nt * 16 + (l & 15);
    int k = 32 * c + ((l < 16) ? 0 : 8) + i + ((i >= 8) ? 8 : 0);
    const float* W = (mat == 0) ? Wa : ((mat == 1) ? Wp1 : Wp2);
    pk1[idx] = f2bf(W[(size_t)(1 + k) * HH + n]);   // skip scalar row 0
  } else if (idx < S1_HALVES + S2_HALVES) {
    int j = idx - S1_HALVES;
    int i  =  j        & 15;
    int l  = (j >> 4)  & 31;
    int c  = (j >> 9)  & 15;
    int nt = (j >> 13) & 15;
    int mat = j >> 17;
    int n = nt * 16 + (l & 15);
    int k = 32 * c + ((l < 16) ? 0 : 8) + i + ((i >= 8) ? 8 : 0);
    const float* W = (mat == 0) ? Wz : Wh;
    pk2[j] = f2bf(W[(size_t)k * HH + n]);
  }
}

__global__ void pgj_pack_aux(const float* __restrict__ Wa,
                             const float* __restrict__ Wp1,
                             const float* __restrict__ Wp2,
                             const float* __restrict__ ba,
                             const float* __restrict__ bp1,
                             const float* __restrict__ bp2,
                             const float* __restrict__ bz,
                             const float* __restrict__ bh,
                             const float* __restrict__ Wo,
                             const float* __restrict__ bo,
                             float* __restrict__ aux) {
  int i = blockIdx.x * blockDim.x + threadIdx.x;
  if (i >= AX_N) return;
  float v;
  if      (i < 256)  v = Wa [i];          // row 0 (amp weight)
  else if (i < 512)  v = Wp1[i - 256];    // row 0 (cos weight)
  else if (i < 768)  v = Wp2[i - 512];    // row 0 (sin weight)
  else if (i < 1024) v = ba [i - 768];
  else if (i < 1280) v = bp1[i - 1024];
  else if (i < 1536) v = bp2[i - 1280];
  else if (i < 1792) v = bz [i - 1536];
  else if (i < 2048) v = bh [i - 1792];
  else if (i < 2560) v = Wo [i - 2048];   // (256,2) row-major
  else               v = bo [i - 2560];
  aux[i] = v;
}

// ------------------------------ main kernel -------------------------------
__global__ __launch_bounds__(512, 1)
void pgj_scan(const float* __restrict__ x,      // (B,T,2)
              const float* __restrict__ h0,     // (B,H)
              const ushort16* __restrict__ pk1,
              const ushort16* __restrict__ pk2,
              const float* __restrict__ aux,
              float* __restrict__ out,          // (B,T,2)
              float* __restrict__ hn)           // (B,H)
{
  __shared__ float    sh_h32[16 * P32];
  __shared__ ushort16 sh_h16[16 * P16];
  __shared__ ushort16 sh_u16[16 * P16];
  __shared__ float    sh_amp[16], sh_cos[16], sh_sin[16];
  __shared__ float    sh_aux[AX_N];
  __shared__ float    sh_part[16][16][2];
  // LDS-resident slice of packed Wz: [ntile(16)][c(<7)][lane(32)][i(16)]
  __shared__ __align__(16) ushort16 sh_wz[16 * WZ_LDS_CHUNKS * 32 * 16];

  const int tid   = threadIdx.x;
  const int lane  = tid & 31;
  const int w     = tid >> 5;              // wave id == N-tile (0..15)
  const int lm    = lane & 15;
  const int mhalf = (lane < 16) ? 0 : 8;   // C/D fragment: m = r + mhalf
  const int kbase = (lane < 16) ? 0 : 8;   // A fragment K sub-offset
  const int nLoc  = w * 16 + lm;           // this lane's output column
  const int btile = blockIdx.x * 16;

  // ---- TDM: stage Wz K-chunks 0..6 of every ntile into LDS (wave 0) ----
  // Source in pk2 (mat 0): per ntile a contiguous run of 7*512 = 3584 halves
  // at stride 16*512 = 8192 halves -> 2D tensor: 16 rows x 3584, stride 8192.
  if (tid < 32) {
    const uint32 lds_off = (uint32)(size_t)(void*)sh_wz;   // low 32 bits = LDS byte offset
    const unsigned long long ga = (unsigned long long)(size_t)pk2;
    u32x4 g0;
    g0[0] = 1u;                                            // count=1, user D#
    g0[1] = lds_off;                                       // lds_addr
    g0[2] = (uint32)(ga & 0xFFFFFFFFu);                    // global_addr[31:0]
    g0[3] = (uint32)((ga >> 32) & 0x01FFFFFFu)             // global_addr[56:32]
            | 0x80000000u;                                 // type=2 (image)
    i32x8 g1;
    g1[0] = 0x00010000;        // workgroup_mask=0, data_size=1 (2 bytes)
    g1[1] = (int)0x0E000000u;  // tensor_dim0 = 3584 (lo16 in bits [31:16])
    g1[2] = (int)0x00100000u;  // tensor_dim1 = 16   (lo16 in bits [31:16])
    g1[3] = (int)0x0E000000u;  // tile_dim0   = 3584 (bits [31:16])
    g1[4] = 16;                // tile_dim1 = 16, tile_dim2 = 0
    g1[5] = 8192;              // tensor_dim0_stride[31:0] (data_size units)
    g1[6] = 0;                 // stride0 hi, stride1 lo
    g1[7] = 0;                 // stride1 hi
    i32x4 z4 = {0, 0, 0, 0};
    i32x8 z8 = {0, 0, 0, 0, 0, 0, 0, 0};
    __builtin_amdgcn_tensor_load_to_lds(g0, g1, z4, z4, z8, 0);
    __builtin_amdgcn_s_wait_tensorcnt(0);
  }

  for (int i = tid; i < AX_N; i += 512) sh_aux[i] = aux[i];
  for (int e = tid; e < 16 * HH; e += 512) {
    int m = e >> 8, n = e & 255;
    float v = h0[(size_t)(btile + m) * HH + n];
    sh_h32[m * P32 + n] = v;
    sh_h16[m * P16 + n] = f2bf(v);
  }

  const ushort16* arow = sh_h16 + lm * P16;
  const ushort16* urow = sh_u16 + lm * P16;

  for (int tt = 0; tt < TT; ++tt) {
    __syncthreads();                       // (1) h16/h32/u16/LDS-weights valid
    if (w == 0 && lane < 16) {
      const float2 xv = *(const float2*)(x + ((size_t)(btile + lane) * TT + tt) * 2);
      sh_amp[lane] = xv.x;
      sh_cos[lane] = __cosf(xv.y);
      sh_sin[lane] = __sinf(xv.y);
    }

    // ---- stage 1: a/p1/p2 = tanh(h @ W[1:] + scalar*W[0] + b), K=256 ----
    v8f ca = {}, cp1 = {}, cp2 = {};
#pragma unroll
    for (int c = 0; c < 8; ++c) {
      v16bf a  = load_afrag(arow, 32 * c + kbase);
      v16bf b0 = load_bfrag(pk1 + (((0 * 16 + w) * 8 + c) * 32 + lane) * 16);
      v16bf b1 = load_bfrag(pk1 + (((1 * 16 + w) * 8 + c) * 32 + lane) * 16);
      v16bf b2 = load_bfrag(pk1 + (((2 * 16 + w) * 8 + c) * 32 + lane) * 16);
      ca  = wmma_bf16(a, b0, ca);
      cp1 = wmma_bf16(a, b1, cp1);
      cp2 = wmma_bf16(a, b2, cp2);
    }
    __syncthreads();                       // (2) amp/cos/sin visible

    const float wa0 = sh_aux[AX_WA0  + nLoc];
    const float wp0 = sh_aux[AX_WP10 + nLoc];
    const float ws0 = sh_aux[AX_WP20 + nLoc];
    const float vba = sh_aux[AX_BA  + nLoc];
    const float vb1 = sh_aux[AX_BP1 + nLoc];
    const float vb2 = sh_aux[AX_BP2 + nLoc];
#pragma unroll
    for (int r = 0; r < 8; ++r) {
      int m = r + mhalf;
      float a  = fast_tanh(ca[r]  + vba + sh_amp[m] * wa0);
      float p1 = fast_tanh(cp1[r] + vb1 + sh_cos[m] * wp0);
      float p2 = fast_tanh(cp2[r] + vb2 + sh_sin[m] * ws0);
      float u  = a * p1 * p2 * (1.f - a) * (1.f - p1) * (1.f - p2);
      sh_u16[m * P16 + nLoc] = f2bf(u);
    }
    __syncthreads();                       // (3) u16 ready

    // ---- stage 2: z/hc over newin=[u,h], K=512 ----
    v8f cz = {}, chc = {};
#pragma unroll
    for (int c = 0; c < 16; ++c) {
      v16bf a = (c < 8) ? load_afrag(urow, 32 * c + kbase)
                        : load_afrag(arow, 32 * (c - 8) + kbase);
      v16bf b0 = (c < WZ_LDS_CHUNKS)
                   ? load_bfrag(sh_wz + ((w * WZ_LDS_CHUNKS + c) * 32 + lane) * 16)
                   : load_bfrag(pk2 + (((0 * 16 + w) * 16 + c) * 32 + lane) * 16);
      v16bf b1 = load_bfrag(pk2 + (((1 * 16 + w) * 16 + c) * 32 + lane) * 16);
      cz  = wmma_bf16(a, b0, cz);
      chc = wmma_bf16(a, b1, chc);
    }

    const float vbz = sh_aux[AX_BZ + nLoc];
    const float vbh = sh_aux[AX_BH + nLoc];
    float hnew[8];
#pragma unroll
    for (int r = 0; r < 8; ++r) {
      int m = r + mhalf;
      float z  = 0.5f * fast_tanh(0.5f * (cz[r] + vbz)) + 0.5f;  // sigmoid
      float hc = fast_tanh(chc[r] + vbh);
      float ho = sh_h32[m * P32 + nLoc];
      hnew[r] = z * ho + (1.f - z) * hc;
    }
    __syncthreads();                       // (4) all reads of old h/u done

    const float wo0 = sh_aux[AX_WO + nLoc * 2 + 0];
    const float wo1 = sh_aux[AX_WO + nLoc * 2 + 1];
#pragma unroll
    for (int r = 0; r < 8; ++r) {
      int m = r + mhalf;
      sh_h32[m * P32 + nLoc] = hnew[r];
      sh_h16[m * P16 + nLoc] = f2bf(hnew[r]);
      // inline hs@Wo partial: reduce over n within each 16-lane half
      float p0 = hnew[r] * wo0;
      float p1 = hnew[r] * wo1;
#pragma unroll
      for (int msk = 1; msk <= 8; msk <<= 1) {
        p0 += __shfl_xor(p0, msk, 32);
        p1 += __shfl_xor(p1, msk, 32);
      }
      if (lm == 0) { sh_part[w][m][0] = p0; sh_part[w][m][1] = p1; }
    }
    __syncthreads();                       // (5) partials ready

    if (w == 0) {
      int m = lane >> 1, cc = lane & 1;
      float s = sh_aux[AX_BO + cc];
#pragma unroll
      for (int ww = 0; ww < 16; ++ww) s += sh_part[ww][m][cc];
      out[((size_t)(btile + m) * TT + tt) * 2 + cc] = s;
    }
  }

  __syncthreads();
  for (int e = tid; e < 16 * HH; e += 512) {
    int m = e >> 8, n = e & 255;
    hn[(size_t)(btile + m) * HH + n] = sh_h32[m * P32 + n];
  }
}

extern "C" void kernel_launch(void* const* d_in, const int* in_sizes, int n_in,
                              void* d_out, int out_size, void* d_ws, size_t ws_size,
                              hipStream_t stream) {
  const float* x   = (const float*)d_in[0];
  const float* h0  = (const float*)d_in[1];
  const float* Wa  = (const float*)d_in[2];
  const float* ba  = (const float*)d_in[3];
  const float* Wp1 = (const float*)d_in[4];
  const float* bp1 = (const float*)d_in[5];
  const float* Wp2 = (const float*)d_in[6];
  const float* bp2 = (const float*)d_in[7];
  const float* Wz  = (const float*)d_in[8];
  const float* bz  = (const float*)d_in[9];
  const float* Wh  = (const float*)d_in[10];
  const float* bh  = (const float*)d_in[11];
  const float* Wo  = (const float*)d_in[12];
  const float* bo  = (const float*)d_in[13];

  ushort16* pk1 = (ushort16*)d_ws;
  ushort16* pk2 = pk1 + S1_HALVES;
  float*    aux = (float*)(pk2 + S2_HALVES);

  float* out = (float*)d_out;
  float* hn  = out + (size_t)BB * TT * 2;

  pgj_pack_weights<<<(S1_HALVES + S2_HALVES + 255) / 256, 256, 0, stream>>>(
      Wa, Wp1, Wp2, Wz, Wh, pk1, pk2);
  pgj_pack_aux<<<(AX_N + 255) / 256, 256, 0, stream>>>(
      Wa, Wp1, Wp2, ba, bp1, bp2, bz, bh, Wo, bo, aux);
  pgj_scan<<<BB / 16, 512, 0, stream>>>(x, h0, pk1, pk2, aux, out, hn);
}
